// FiLMGRU_7043746365710
// MI455X (gfx1250) — matrix operand order repaired and checked by reading, not compiled, not run on hardware
//
#include <hip/hip_runtime.h>
#include <hip/hip_bf16.h>
#include <math.h>
#include <stdint.h>

// ---------------------------------------------------------------------------
// FiLM-GRU, 2 layers, fused WMMA step kernel for gfx1250 (CDNA5, wave32).
//   T=128, B=1024, IN=128, H=256, gates G=3H=768.
// v2: block = 64(b) x 32(h) macro-tile (8 waves = 4 b-tiles x 2 g-tiles).
//     All B (weight) panels for the block are staged once into LDS with
//     CDNA5 async global->LDS copies; waves then feed WMMA from LDS.
//     This cuts per-step weight traffic 4x vs per-wave global reads.
// ---------------------------------------------------------------------------

#define T_STEPS 128
#define BATCH   1024
#define IN_DIM  128
#define H_DIM   256
#define G_DIM   (3 * H_DIM)          // 768

#define LDS_COL   40                 // bf16 per column: 32 data + 8 pad (bank-safe)
#define PANEL_EL  (16 * LDS_COL)     // 640 bf16 = 1280 B per (chunk,gate,gtile) panel

typedef __attribute__((ext_vector_type(16))) __bf16 v16bf;
typedef __attribute__((ext_vector_type(8)))  __bf16 v8bf;
typedef __attribute__((ext_vector_type(8)))  float  v8f;

__device__ __forceinline__ v8f wmma_bf16(v16bf a, v16bf b, v8f c) {
    return __builtin_amdgcn_wmma_f32_16x16x32_bf16(
        false, a, false, b, (short)0, c, false, false);
}

// A fragment (16x32 bf16 activations, row-major). Lane L: row M = L%16.
// Contiguous K layout (matches B): element s holds K{base+s}, s = 0..15,
// base = (lane<16 ? 0 : 16).
__device__ __forceinline__ v16bf load_a_frag(const __bf16* __restrict__ p, int hi) {
    const v8bf lo  = *(const v8bf*)(p + hi * 16);
    const v8bf hi8 = *(const v8bf*)(p + hi * 16 + 8);
    v16bf a;
#pragma unroll
    for (int i = 0; i < 8; ++i) { a[i] = lo[i]; a[8 + i] = hi8[i]; }
    return a;
}

// B fragment from LDS panel: lane L = col n (L%16), 16 contiguous K values
// starting at kk = hi*16. Column stride LDS_COL keeps 16B alignment and
// distinct banks across a half-wave.
__device__ __forceinline__ v16bf load_b_lds(const __bf16* lp) {
    const v8bf a0 = *(const v8bf*)(lp);
    const v8bf a1 = *(const v8bf*)(lp + 8);
    v16bf b;
#pragma unroll
    for (int i = 0; i < 8; ++i) { b[i] = a0[i]; b[8 + i] = a1[i]; }
    return b;
}

__device__ __forceinline__ float sigmoidf_fast(float x) {
    return 1.0f / (1.0f + __expf(-x));
}

// ---------------------------------------------------------------------------
// Fused GRU step. grid = (B/64)*(H/32) = 128 blocks of 256 threads.
// Dynamic LDS: nC * 6 panels of PANEL_EL bf16.
// ---------------------------------------------------------------------------
__global__ __launch_bounds__(256)
void film_gru_step(const __bf16* __restrict__ xsrc, int Kx, int nCX, int nC,
                   const __bf16* __restrict__ h_in_bf,
                   const float*  __restrict__ h_in_f,
                   __bf16* __restrict__ h_out_bf,
                   float*  __restrict__ h_out_f,
                   const __bf16* __restrict__ BpX,
                   const __bf16* __restrict__ BpH,
                   const float* __restrict__ bx, const float* __restrict__ bh,
                   const float* __restrict__ gamma_, const float* __restrict__ beta_,
                   float*  __restrict__ out_f,
                   __bf16* __restrict__ out_bf,
                   float*  __restrict__ state_out)
{
    extern __shared__ __bf16 ldsB[];

    const int lane = threadIdx.x & 31;
    const int wave = threadIdx.x >> 5;
    const int bblk = (blockIdx.x >> 3) * 64;      // H/32 = 8 g-blocks per b-row
    const int gblk = (blockIdx.x & 7) * 32;
    const int bt = wave & 3;                      // 4 b-tiles per block
    const int gt = wave >> 2;                     // 2 g-tiles per block
    const int b0 = bblk + bt * 16;
    const int g0 = gblk + gt * 16;
    const int nIdx = lane & 15;
    const int hi   = lane >> 4;

    // ---- stage all B panels for this block into LDS (async copies) ------
    // Panel p = (c*3 + e)*2 + gt2 ; 16 cols x 32 K-values, col stride LDS_COL.
    {
        const uint32_t ldsBase = (uint32_t)(uintptr_t)ldsB;
        const int tasks = nC * 6 * 64;            // 16 cols * 4 x 16B per panel
        for (int tsk = threadIdx.x; tsk < tasks; tsk += 256) {
            const int q   = tsk & 3;
            const int col = (tsk >> 2) & 15;
            const int p   = tsk >> 6;
            const int gt2 = p & 1;
            const int e   = (p >> 1) % 3;
            const int c   = p / 6;
            const __bf16* src = (c < nCX)
                ? BpX + (((size_t)c * G_DIM) << 5)
                : BpH + (((size_t)(c - nCX) * G_DIM) << 5);
            src += ((size_t)(e * H_DIM + gblk + gt2 * 16 + col) << 5) + (q << 3);
            const uint32_t dst = ldsBase + (uint32_t)p * (PANEL_EL * 2)
                               + col * (LDS_COL * 2) + (q << 4);
            const uint64_t ga = (uint64_t)(uintptr_t)src;
            asm volatile("global_load_async_to_lds_b128 %0, %1, off"
                         :: "v"(dst), "v"(ga) : "memory");
        }
        asm volatile("s_wait_asynccnt 0x0" ::: "memory");
        __syncthreads();
    }

    const int arow = b0 + nIdx;                   // A-matrix row for this lane

    v8f accZ = {}, accR = {}, accXN = {}, accHN = {};

    // ---- x-side chunks (c = 0..nCX-1) ------------------------------------
    const __bf16* aX = xsrc + (size_t)arow * Kx;
    const __bf16* lbase = ldsB + nIdx * LDS_COL + hi * 16;
    for (int c = 0; c < nCX; ++c) {
        __builtin_prefetch(aX + (c + 1) * 32, 0, 1);
        const v16bf A  = load_a_frag(aX + c * 32, hi);
        const __bf16* lp = lbase + ((c * 3) * 2 + gt) * PANEL_EL;
        const v16bf Bz = load_b_lds(lp);
        const v16bf Br = load_b_lds(lp + 2 * PANEL_EL);
        const v16bf Bn = load_b_lds(lp + 4 * PANEL_EL);
        accZ  = wmma_bf16(A, Bz, accZ);
        accR  = wmma_bf16(A, Br, accR);
        accXN = wmma_bf16(A, Bn, accXN);
    }

    // ---- h-side chunks (c = nCX..nC-1) -----------------------------------
    const __bf16* aH = h_in_bf + (size_t)arow * H_DIM;
#pragma unroll 4
    for (int c = nCX; c < nC; ++c) {
        const v16bf A  = load_a_frag(aH + (c - nCX) * 32, hi);
        const __bf16* lp = lbase + ((c * 3) * 2 + gt) * PANEL_EL;
        const v16bf Bz = load_b_lds(lp);
        const v16bf Br = load_b_lds(lp + 2 * PANEL_EL);
        const v16bf Bn = load_b_lds(lp + 4 * PANEL_EL);
        accZ  = wmma_bf16(A, Bz, accZ);
        accR  = wmma_bf16(A, Br, accR);
        accHN = wmma_bf16(A, Bn, accHN);
    }

    // ---- gates + FiLM (C/D layout: VGPR r -> row r + hi*8, col = nIdx) ---
    const int g = g0 + nIdx;
    const float bz  = bx[g]             + bh[g];
    const float br  = bx[H_DIM + g]     + bh[H_DIM + g];
    const float bnx = bx[2 * H_DIM + g];
    const float bnh = bh[2 * H_DIM + g];

#pragma unroll
    for (int r = 0; r < 8; ++r) {
        const int row = b0 + r + hi * 8;
        const int idx = row * H_DIM + g;
        const float z  = sigmoidf_fast(accZ[r] + bz);
        const float rr = sigmoidf_fast(accR[r] + br);
        const float n  = tanhf(accXN[r] + bnx + rr * (accHN[r] + bnh));
        const float hprev = h_in_f[idx];
        float hnew = fmaf(z, hprev - n, n);              // (1-z)n + z*h
        hnew = fmaf(gamma_[idx], hnew, beta_[idx]);      // FiLM
        h_out_f[idx]  = hnew;
        h_out_bf[idx] = (__bf16)hnew;
        if (out_f)     out_f[idx]     = hnew;
        if (out_bf)    out_bf[idx]    = (__bf16)hnew;
        if (state_out) state_out[idx] = hnew;
    }
}

// ---------------------------------------------------------------------------
// Prologue kernels
// ---------------------------------------------------------------------------
__global__ void convert_f32_to_bf16(const float* __restrict__ src,
                                    __bf16* __restrict__ dst, int n) {
    int i = blockIdx.x * blockDim.x + threadIdx.x;
    if (i < n) dst[i] = (__bf16)src[i];
}

// Pack W [G, K] (row-major fp32) into B-operand panels:
// Bpack[(c*G + g)*32 + kk] = (bf16) W[g*K + c*32 + kk]
__global__ void pack_weights(const float* __restrict__ W,
                             __bf16* __restrict__ Bp, int K) {
    int i = blockIdx.x * blockDim.x + threadIdx.x;
    if (i >= K * G_DIM) return;
    int g = i / K;
    int k = i - g * K;
    Bp[(((size_t)(k >> 5) * G_DIM + g) << 5) + (k & 31)] = (__bf16)W[(size_t)g * K + k];
}

__global__ void init_hidden(const float* __restrict__ s0,
                            float* __restrict__ hf0, __bf16* __restrict__ hb0,
                            float* __restrict__ hf1, __bf16* __restrict__ hb1) {
    int i = blockIdx.x * blockDim.x + threadIdx.x;
    const int n = BATCH * H_DIM;
    if (i < n) {
        float a = s0[i];
        float b = s0[n + i];
        hf0[i] = a; hb0[i] = (__bf16)a;
        hf1[i] = b; hb1[i] = (__bf16)b;
    }
}

// ---------------------------------------------------------------------------
extern "C" void kernel_launch(void* const* d_in, const int* in_sizes, int n_in,
                              void* d_out, int out_size, void* d_ws, size_t ws_size,
                              hipStream_t stream) {
    const float* x    = (const float*)d_in[0];
    const float* init = (const float*)d_in[1];
    const float* gam  = (const float*)d_in[2];
    const float* bet  = (const float*)d_in[3];
    const float* Wx0  = (const float*)d_in[4];
    const float* bx0  = (const float*)d_in[5];
    const float* Wh0  = (const float*)d_in[6];
    const float* bh0  = (const float*)d_in[7];
    const float* Wx1  = (const float*)d_in[8];
    const float* bx1  = (const float*)d_in[9];
    const float* Wh1  = (const float*)d_in[10];
    const float* bh1  = (const float*)d_in[11];
    float* out = (float*)d_out;

    // ---- workspace carve-up (256B aligned chunks) ------------------------
    char* ws = (char*)d_ws;
    size_t off = 0;
    auto carve = [&](size_t bytes) -> char* {
        char* p = ws + off;
        off = (off + bytes + 255) & ~(size_t)255;
        return p;
    };
    __bf16* x_bf    = (__bf16*)carve((size_t)T_STEPS * BATCH * IN_DIM * 2);
    __bf16* out0_bf = (__bf16*)carve((size_t)T_STEPS * BATCH * H_DIM * 2);
    __bf16* BpX0 = (__bf16*)carve((size_t)IN_DIM * G_DIM * 2);
    __bf16* BpH0 = (__bf16*)carve((size_t)H_DIM  * G_DIM * 2);
    __bf16* BpX1 = (__bf16*)carve((size_t)H_DIM  * G_DIM * 2);
    __bf16* BpH1 = (__bf16*)carve((size_t)H_DIM  * G_DIM * 2);
    const size_t HS = (size_t)BATCH * H_DIM;
    float*  hf[2][2];  __bf16* hb[2][2];
    for (int l = 0; l < 2; ++l)
        for (int p = 0; p < 2; ++p) {
            hf[l][p] = (float*)carve(HS * 4);
            hb[l][p] = (__bf16*)carve(HS * 2);
        }

    // Allow >48KB dynamic LDS for the step kernel (host-side, capture-safe).
    hipFuncSetAttribute((const void*)film_gru_step,
                        hipFuncAttributeMaxDynamicSharedMemorySize,
                        16 * 6 * PANEL_EL * 2);

    // ---- prologue --------------------------------------------------------
    {
        int n = T_STEPS * BATCH * IN_DIM;
        convert_f32_to_bf16<<<(n + 255) / 256, 256, 0, stream>>>(x, x_bf, n);
    }
    pack_weights<<<(IN_DIM * G_DIM + 255) / 256, 256, 0, stream>>>(Wx0, BpX0, IN_DIM);
    pack_weights<<<(H_DIM * G_DIM + 255) / 256, 256, 0, stream>>>(Wh0, BpH0, H_DIM);
    pack_weights<<<(H_DIM * G_DIM + 255) / 256, 256, 0, stream>>>(Wx1, BpX1, H_DIM);
    pack_weights<<<(H_DIM * G_DIM + 255) / 256, 256, 0, stream>>>(Wh1, BpH1, H_DIM);
    init_hidden<<<((int)HS + 255) / 256, 256, 0, stream>>>(
        init, hf[0][0], hb[0][0], hf[1][0], hb[1][0]);

    const int stepBlocks = (BATCH / 64) * (H_DIM / 32);       // 128
    float* state0 = out + (size_t)T_STEPS * BATCH * H_DIM;
    float* state1 = state0 + HS;

    const int nCX0 = IN_DIM / 32, nC0 = (IN_DIM + H_DIM) / 32;   // 4, 12
    const int nCX1 = H_DIM / 32,  nC1 = (2 * H_DIM) / 32;        // 8, 16
    const size_t sh0 = (size_t)nC0 * 6 * PANEL_EL * 2;           // 92160 B
    const size_t sh1 = (size_t)nC1 * 6 * PANEL_EL * 2;           // 122880 B

    // ---- layer 0 ---------------------------------------------------------
    for (int t = 0; t < T_STEPS; ++t) {
        int pi = t & 1, po = (t + 1) & 1;
        film_gru_step<<<stepBlocks, 256, sh0, stream>>>(
            x_bf + (size_t)t * BATCH * IN_DIM, IN_DIM, nCX0, nC0,
            hb[0][pi], hf[0][pi], hb[0][po], hf[0][po],
            BpX0, BpH0, bx0, bh0, gam, bet,
            nullptr, out0_bf + (size_t)t * BATCH * H_DIM,
            (t == T_STEPS - 1) ? state0 : nullptr);
    }

    // ---- layer 1 ---------------------------------------------------------
    for (int t = 0; t < T_STEPS; ++t) {
        int pi = t & 1, po = (t + 1) & 1;
        film_gru_step<<<stepBlocks, 256, sh1, stream>>>(
            out0_bf + (size_t)t * BATCH * H_DIM, H_DIM, nCX1, nC1,
            hb[1][pi], hf[1][pi], hb[1][po], hf[1][po],
            BpX1, BpH1, bx1, bh1, gam + HS, bet + HS,
            out + (size_t)t * BATCH * H_DIM, nullptr,
            (t == T_STEPS - 1) ? state1 : nullptr);
    }
    (void)in_sizes; (void)n_in; (void)out_size; (void)ws_size;
}
